// VolumeTransform_70798240907494
// MI455X (gfx1250) — compile-verified, matrix operand order checked by reading, not run on hardware
//
#include <hip/hip_runtime.h>

// ---------------- types / helpers ----------------
typedef __attribute__((ext_vector_type(16))) __bf16 v16bf;
typedef __attribute__((ext_vector_type(8)))  float  v8f;
typedef __attribute__((ext_vector_type(4)))  unsigned int u32x4;

union FragAB { v16bf v; u32x4 u4[2]; };

#define DEV __device__ __forceinline__

DEV unsigned short f2bf(float f) {
    unsigned int u = __float_as_uint(f);
    u += 0x7FFFu + ((u >> 16) & 1u);          // round-to-nearest-even
    return (unsigned short)(u >> 16);
}
DEV float bf2f(unsigned short h) {
    return __uint_as_float(((unsigned int)h) << 16);
}
DEV v8f wmma_bf16(const FragAB& a, const FragAB& b, v8f c) {
    return __builtin_amdgcn_wmma_f32_16x16x32_bf16(false, a.v, false, b.v,
                                                   (short)0, c, false, false);
}

// problem constants
#define NQ_    20000
#define D_     256
#define FFN_   512
#define CAMS_  6
#define LPIX_  176   // 8*22
#define FH_    8
#define FW_    22
#define INC_   768
#define NVOX_  160000            // 100*100*16
#define XOUT_  30720000          // 192*160000

constexpr int F_RELU  = 1;
constexpr int F_WF32  = 2;
constexpr int F_WBF16 = 4;
constexpr int F_EMB   = 8;

// ---------------- packing / casting kernels ----------------
__global__ void k_cast_bf16(const float* __restrict__ s, unsigned short* __restrict__ d, int n) {
    int i = blockIdx.x * blockDim.x + threadIdx.x;
    if (i < n) d[i] = f2bf(s[i]);
}

// src fp32 [K][N]  ->  dst bf16 [N][K]
__global__ void k_pack_bt(const float* __restrict__ s, unsigned short* __restrict__ d, int K, int N) {
    int i = blockIdx.x * blockDim.x + threadIdx.x;
    if (i >= K * N) return;
    int kk = i % K, nn = i / K;
    d[nn * K + kk] = f2bf(s[kk * N + nn]);
}

// deconv_w (256,256,2,2,2) -> bf16 [8][256 oc][256 ic]
__global__ void k_pack_deconv(const float* __restrict__ w, unsigned short* __restrict__ d) {
    int i = blockIdx.x * blockDim.x + threadIdx.x;
    if (i >= 8 * 256 * 256) return;
    int ic = i % 256, oc = (i / 256) % 256, t = i / 65536;
    d[i] = f2bf(w[(oc * 256 + ic) * 8 + t]);
}

// conv3_w (192,256,3,3,3) -> bf16 [27][192 oc][256 ic]
__global__ void k_pack_conv3(const float* __restrict__ w, unsigned short* __restrict__ d) {
    int i = blockIdx.x * blockDim.x + threadIdx.x;
    if (i >= 27 * 192 * 256) return;
    int ic = i % 256, oc = (i / 256) % 192, t = i / (256 * 192);
    d[i] = f2bf(w[(oc * 256 + ic) * 27 + t]);
}

// unmask/restore: write cam_x output slice + bf16 A matrix [1056][768]
__global__ void k_restore(const float* __restrict__ camx, const int* __restrict__ ids,
                          const float* __restrict__ mask, unsigned short* __restrict__ fullb,
                          float* __restrict__ camx_out) {
    int i = blockIdx.x * blockDim.x + threadIdx.x;
    if (i >= CAMS_ * LPIX_ * INC_) return;
    int c = i % INC_, pix = (i / INC_) % LPIX_, n = i / (INC_ * LPIX_);
    int j = ids[n * LPIX_ + pix];
    float v = (j < 44) ? camx[(n * 44 + j) * INC_ + c] : mask[c];
    camx_out[((size_t)n * INC_ + c) * LPIX_ + pix] = v;
    fullb[((size_t)n * LPIX_ + pix) * INC_ + c]    = f2bf(v);
}

__global__ void k_castq(const float* __restrict__ s, float* __restrict__ qf,
                        unsigned short* __restrict__ qb, int n) {
    int i = blockIdx.x * blockDim.x + threadIdx.x;
    if (i < n) { float v = s[i]; qf[i] = v; qb[i] = f2bf(v); }
}

// ---------------- generic WMMA bf16 GEMM, 32x32 tile per wave ----------------
DEV void gemm_epilogue(v8f acc, int m0, int half, int n, int N, float bv, int flags,
                       const float* __restrict__ resid, float* __restrict__ Cf,
                       unsigned short* __restrict__ Cbf,
                       const float* __restrict__ cam_emb, const float* __restrict__ lvl_emb) {
    #pragma unroll
    for (int j = 0; j < 8; ++j) {
        int m = m0 + j + (half ? 8 : 0);
        float v = acc[j] + bv;
        if (flags & F_RELU) v = fmaxf(v, 0.f);
        if (flags & F_EMB)  v += cam_emb[(m / LPIX_) * D_ + n] + lvl_emb[n];
        if (resid)          v += resid[(size_t)m * N + n];
        if (flags & F_WF32)  Cf [(size_t)m * N + n] = v;
        if (flags & F_WBF16) Cbf[(size_t)m * N + n] = f2bf(v);
    }
}

__global__ __launch_bounds__(32)
void k_gemm(const unsigned short* __restrict__ A, const unsigned short* __restrict__ Bt,
            const float* __restrict__ bias, const float* __restrict__ resid,
            float* __restrict__ Cf, unsigned short* __restrict__ Cbf,
            const float* __restrict__ cam_emb, const float* __restrict__ lvl_emb,
            int M, int K, int N, int flags) {
    int mt = blockIdx.x, nt = blockIdx.y;
    int lane = threadIdx.x;
    int half = lane >> 4, r = lane & 15;
    int klo = half ? 8 : 0;
    const unsigned short* arow0 = A  + (size_t)(mt * 32 + r) * K;
    const unsigned short* arow1 = arow0 + (size_t)16 * K;
    const unsigned short* brow0 = Bt + (size_t)(nt * 32 + r) * K;
    const unsigned short* brow1 = brow0 + (size_t)16 * K;
    v8f c00 = {0.f,0.f,0.f,0.f,0.f,0.f,0.f,0.f};
    v8f c01 = c00, c10 = c00, c11 = c00;
    for (int kb = 0; kb < K; kb += 32) {
        FragAB a0, a1, b0, b1;
        a0.u4[0] = *(const u32x4*)(arow0 + kb + klo);
        a0.u4[1] = *(const u32x4*)(arow0 + kb + klo + 16);
        a1.u4[0] = *(const u32x4*)(arow1 + kb + klo);
        a1.u4[1] = *(const u32x4*)(arow1 + kb + klo + 16);
        b0.u4[0] = *(const u32x4*)(brow0 + kb + klo);
        b0.u4[1] = *(const u32x4*)(brow0 + kb + klo + 16);
        b1.u4[0] = *(const u32x4*)(brow1 + kb + klo);
        b1.u4[1] = *(const u32x4*)(brow1 + kb + klo + 16);
        c00 = wmma_bf16(a0, b0, c00);
        c01 = wmma_bf16(a0, b1, c01);
        c10 = wmma_bf16(a1, b0, c10);
        c11 = wmma_bf16(a1, b1, c11);
    }
    int n0 = nt * 32 + r, n1 = n0 + 16;
    float bv0 = bias ? bias[n0] : 0.f;
    float bv1 = bias ? bias[n1] : 0.f;
    gemm_epilogue(c00, mt * 32,      half, n0, N, bv0, flags, resid, Cf, Cbf, cam_emb, lvl_emb);
    gemm_epilogue(c01, mt * 32,      half, n1, N, bv1, flags, resid, Cf, Cbf, cam_emb, lvl_emb);
    gemm_epilogue(c10, mt * 32 + 16, half, n0, N, bv0, flags, resid, Cf, Cbf, cam_emb, lvl_emb);
    gemm_epilogue(c11, mt * 32 + 16, half, n1, N, bv1, flags, resid, Cf, Cbf, cam_emb, lvl_emb);
}

// ---------------- softmax over groups of 12 ----------------
__global__ void k_softmax12(float* __restrict__ att) {
    int i = blockIdx.x * blockDim.x + threadIdx.x;
    if (i >= NQ_ * 8) return;
    float* p = att + (size_t)(i >> 3) * 96 + (i & 7) * 12;
    float mx = p[0];
    for (int k = 1; k < 12; ++k) mx = fmaxf(mx, p[k]);
    float e[12], s = 0.f;
    for (int k = 0; k < 12; ++k) { e[k] = expf(p[k] - mx); s += e[k]; }
    float inv = 1.f / s;
    for (int k = 0; k < 12; ++k) p[k] = e[k] * inv;
}

// ---------------- deformable bilinear sampling + head aggregation ----------------
__global__ __launch_bounds__(32)
void k_sample(const float* __restrict__ off, const float* __restrict__ att,
              const float* __restrict__ val, unsigned short* __restrict__ agg) {
    int n = blockIdx.x, h = blockIdx.y, lane = threadIdx.x;   // lane = head-dim
    int wq = n % 50, hq = (n / 50) % 50;
    float refx = (wq + 0.5f) * ((float)FW_ / 50.f) - 0.5f;
    float refy = (hq + 0.5f) * ((float)FH_ / 50.f) - 0.5f;
    float acc = 0.f;
    for (int c = 0; c < CAMS_; ++c) {
        const float* vb = val + ((size_t)c * LPIX_) * D_ + h * 32 + lane;
        for (int p = 0; p < 2; ++p) {
            int ob = n * 192 + ((h * CAMS_ + c) * 2 + p) * 2;
            float x = refx + off[ob + 0];
            float y = refy + off[ob + 1];
            float aw = att[(size_t)n * 96 + h * 12 + c * 2 + p];
            float x0 = floorf(x), y0 = floorf(y);
            float wx = x - x0, wy = y - y0;
            int x0i = (int)x0, y0i = (int)y0;
            auto G = [&](int yi, int xi) -> float {
                int yc = yi < 0 ? 0 : (yi > FH_ - 1 ? FH_ - 1 : yi);
                int xc = xi < 0 ? 0 : (xi > FW_ - 1 ? FW_ - 1 : xi);
                return vb[(size_t)(yc * FW_ + xc) * D_];
            };
            acc += aw * ((1.f-wx)*(1.f-wy)*G(y0i, x0i)   + wx*(1.f-wy)*G(y0i, x0i+1)
                       + (1.f-wx)*wy     *G(y0i+1, x0i) + wx*wy     *G(y0i+1, x0i+1));
        }
    }
    agg[(size_t)n * D_ + h * 32 + lane] = f2bf(acc);
}

// ---------------- LayerNorm over 256, writes fp32 + bf16 ----------------
__global__ __launch_bounds__(256)
void k_layernorm(const float* __restrict__ X, const float* __restrict__ g,
                 const float* __restrict__ b, float* __restrict__ Yf,
                 unsigned short* __restrict__ Yb) {
    int row = blockIdx.x, tid = threadIdx.x;
    __shared__ float red[256];
    float v = X[(size_t)row * 256 + tid];
    red[tid] = v; __syncthreads();
    for (int s = 128; s > 0; s >>= 1) { if (tid < s) red[tid] += red[tid + s]; __syncthreads(); }
    float mean = red[0] * (1.f / 256.f); __syncthreads();
    float d = v - mean;
    red[tid] = d * d; __syncthreads();
    for (int s = 128; s > 0; s >>= 1) { if (tid < s) red[tid] += red[tid + s]; __syncthreads(); }
    float var = red[0] * (1.f / 256.f);
    float y = d * rsqrtf(var + 1e-5f) * g[tid] + b[tid];
    Yf[(size_t)row * 256 + tid] = y;
    Yb[(size_t)row * 256 + tid] = f2bf(y);
}

// ---------------- transposed conv (8 tap-GEMMs) -> x1 bf16 [256][NVOX] ----------------
__global__ __launch_bounds__(32)
void k_deconv(const unsigned short* __restrict__ Qb, const unsigned short* __restrict__ Wd,
              unsigned short* __restrict__ X1) {
    int mt = blockIdx.x, nt = blockIdx.y, t = blockIdx.z;
    int tz = t >> 2, ty = (t >> 1) & 1, tx = t & 1;
    int lane = threadIdx.x, half = lane >> 4, r = lane & 15, klo = half ? 8 : 0;
    int ma = mt * 16 + r;                          // A row (input voxel, (p,q,r) order)
    int p = ma / 400, qv = (ma / 8) % 50, rr = ma % 8;
    int qrow = rr * 2500 + qv * 50 + p;            // vol = transpose of q
    const unsigned short* arow = Qb + (size_t)qrow * D_;
    const unsigned short* brow = Wd + (size_t)(t * 256 + nt * 16 + r) * D_;
    v8f acc = {0.f,0.f,0.f,0.f,0.f,0.f,0.f,0.f};
    for (int kb = 0; kb < D_; kb += 32) {
        FragAB a, b;
        a.u4[0] = *(const u32x4*)(arow + kb + klo);
        a.u4[1] = *(const u32x4*)(arow + kb + klo + 16);
        b.u4[0] = *(const u32x4*)(brow + kb + klo);
        b.u4[1] = *(const u32x4*)(brow + kb + klo + 16);
        acc = wmma_bf16(a, b, acc);
    }
    int oc = nt * 16 + r;
    #pragma unroll
    for (int j = 0; j < 8; ++j) {
        int m = mt * 16 + j + (half ? 8 : 0);
        int pp = m / 400, qq = (m / 8) % 50, zz = m % 8;
        int P = 2 * pp + tz, Q = 2 * qq + ty, R = 2 * zz + tx;
        X1[(size_t)oc * NVOX_ + (size_t)(P * 100 + Q) * 16 + R] = f2bf(acc[j]);
    }
}

// ---------------- GroupNorm stats ----------------
__global__ void k_zero(float* p, int n) {
    int i = blockIdx.x * blockDim.x + threadIdx.x;
    if (i < n) p[i] = 0.f;
}

__global__ __launch_bounds__(256)
void k_gnstats_bf16(const unsigned short* __restrict__ X, float* __restrict__ st,
                    int cpg, long spatial) {
    int g = blockIdx.y;
    long total = (long)cpg * spatial;
    long base  = (long)g * total;
    float s0 = 0.f, s1 = 0.f;
    for (long i = (long)blockIdx.x * blockDim.x + threadIdx.x; i < total;
         i += (long)gridDim.x * blockDim.x) {
        float v = bf2f(X[base + i]); s0 += v; s1 += v * v;
    }
    __shared__ float r0[256], r1[256];
    int tid = threadIdx.x;
    r0[tid] = s0; r1[tid] = s1; __syncthreads();
    for (int s = 128; s > 0; s >>= 1) {
        if (tid < s) { r0[tid] += r0[tid + s]; r1[tid] += r1[tid + s]; }
        __syncthreads();
    }
    if (tid == 0) { atomicAdd(&st[g * 2], r0[0]); atomicAdd(&st[g * 2 + 1], r1[0]); }
}

__global__ __launch_bounds__(256)
void k_gnstats_f32(const float* __restrict__ X, float* __restrict__ st,
                   int cpg, long spatial) {
    int g = blockIdx.y;
    long total = (long)cpg * spatial;
    long base  = (long)g * total;
    float s0 = 0.f, s1 = 0.f;
    for (long i = (long)blockIdx.x * blockDim.x + threadIdx.x; i < total;
         i += (long)gridDim.x * blockDim.x) {
        float v = X[base + i]; s0 += v; s1 += v * v;
    }
    __shared__ float r0[256], r1[256];
    int tid = threadIdx.x;
    r0[tid] = s0; r1[tid] = s1; __syncthreads();
    for (int s = 128; s > 0; s >>= 1) {
        if (tid < s) { r0[tid] += r0[tid + s]; r1[tid] += r1[tid + s]; }
        __syncthreads();
    }
    if (tid == 0) { atomicAdd(&st[g * 2], r0[0]); atomicAdd(&st[g * 2 + 1], r1[0]); }
}

// normalize+relu x1 [ch][vox] bf16 -> x1n [vox][ch] bf16 (conv A layout)
__global__ void k_gn1norm(const unsigned short* __restrict__ X1, const float* __restrict__ st,
                          const float* __restrict__ g, const float* __restrict__ b,
                          unsigned short* __restrict__ Xn) {
    long i = (long)blockIdx.x * blockDim.x + threadIdx.x;
    if (i >= (long)D_ * NVOX_) return;
    int ch = (int)(i / NVOX_);
    long vox = i % NVOX_;
    int gr = ch >> 4;
    const float cnt = 16.f * (float)NVOX_;
    float m = st[gr * 2] / cnt;
    float var = st[gr * 2 + 1] / cnt - m * m;
    float v = bf2f(X1[i]);
    float y = (v - m) * rsqrtf(var + 1e-5f) * g[ch] + b[ch];
    Xn[vox * D_ + ch] = f2bf(fmaxf(y, 0.f));
}

// in-place GN2 + relu on d_out [192][NVOX]
__global__ void k_gn2norm(float* __restrict__ X, const float* __restrict__ st,
                          const float* __restrict__ g, const float* __restrict__ b) {
    long i = (long)blockIdx.x * blockDim.x + threadIdx.x;
    if (i >= (long)192 * NVOX_) return;
    int ch = (int)(i / NVOX_);
    int gr = ch / 12;
    const float cnt = 12.f * (float)NVOX_;
    float m = st[gr * 2] / cnt;
    float var = st[gr * 2 + 1] / cnt - m * m;
    float y = (X[i] - m) * rsqrtf(var + 1e-5f) * g[ch] + b[ch];
    X[i] = fmaxf(y, 0.f);
}

// ---- 3x3x3 conv 256->192, implicit GEMM; A tile staged via async global->LDS ----
__global__ __launch_bounds__(384)
void k_conv3(const unsigned short* __restrict__ Xn, const unsigned short* __restrict__ Wp,
             float* __restrict__ Out) {
    __shared__ __align__(16) unsigned short At[16][32];
    int mt = blockIdx.x;
    int tid = threadIdx.x;
    int wave = tid >> 5, lane = tid & 31;
    int half = lane >> 4, r = lane & 15, klo = half ? 8 : 0;
    // staging role (threads 0..255): row sm, dword column skk/2
    int sm = tid >> 4, skk = (tid & 15) * 2;
    int vox = mt * 16 + sm;
    int P = vox / 1600, Q = (vox / 16) % 100, R = vox % 16;
    unsigned ldsoff = (unsigned)(unsigned long long)(size_t)&At[sm][skk];
    v8f acc = {0.f,0.f,0.f,0.f,0.f,0.f,0.f,0.f};
    for (int t = 0; t < 27; ++t) {
        int dz = t / 9 - 1, dy = (t / 3) % 3 - 1, dx = t % 3 - 1;
        int Pp = P + dz, Qp = Q + dy, Rp = R + dx;
        bool inb = (Pp >= 0) & (Pp < 100) & (Qp >= 0) & (Qp < 100) & (Rp >= 0) & (Rp < 16);
        long voxp = ((long)Pp * 100 + Qp) * 16 + Rp;
        const unsigned short* bbase = Wp + (size_t)(t * 192 + wave * 16 + r) * D_;
        for (int kb = 0; kb < D_; kb += 32) {
            __syncthreads();
            if (tid < 256) {
                if (inb) {
                    unsigned long long gaddr =
                        (unsigned long long)(size_t)(Xn + voxp * D_ + kb + skk);
                    // CDNA5 async DMA: global -> LDS, tracked by ASYNCcnt
                    asm volatile("global_load_async_to_lds_b32 %0, %1, off"
                                 :: "v"(ldsoff), "v"(gaddr) : "memory");
                } else {
                    *(unsigned int*)&At[sm][skk] = 0u;
                }
            }
            asm volatile("s_wait_asynccnt 0x0" ::: "memory");
            __syncthreads();
            __builtin_prefetch((const void*)(bbase + kb + 32), 0, 1);
            FragAB a, b;
            a.u4[0] = *(const u32x4*)&At[r][klo];
            a.u4[1] = *(const u32x4*)&At[r][klo + 16];
            b.u4[0] = *(const u32x4*)(bbase + kb + klo);
            b.u4[1] = *(const u32x4*)(bbase + kb + klo + 16);
            acc = wmma_bf16(a, b, acc);
        }
    }
    int oc = wave * 16 + r;
    #pragma unroll
    for (int j = 0; j < 8; ++j) {
        int m = mt * 16 + j + (half ? 8 : 0);
        Out[(size_t)oc * NVOX_ + m] = acc[j];
    }
}

// ---------------- host orchestration ----------------
extern "C" void kernel_launch(void* const* d_in, const int* in_sizes, int n_in,
                              void* d_out, int out_size, void* d_ws, size_t ws_size,
                              hipStream_t stream) {
    const float* camera_x     = (const float*)d_in[0];
    const int*   ids          = (const int*)  d_in[1];
    const float* mask_token   = (const float*)d_in[4];
    const float* vol_emb      = (const float*)d_in[5];
    const float* W_tc         = (const float*)d_in[6];
    const float* b_tc         = (const float*)d_in[7];
    const float* cams_embeds  = (const float*)d_in[8];
    const float* level_embeds = (const float*)d_in[9];
    const float* W_off = (const float*)d_in[10];
    const float* b_off = (const float*)d_in[11];
    const float* W_att = (const float*)d_in[12];
    const float* b_att = (const float*)d_in[13];
    const float* W_val = (const float*)d_in[14];
    const float* b_val = (const float*)d_in[15];
    const float* W_out = (const float*)d_in[16];
    const float* b_out = (const float*)d_in[17];
    const float* ln1_g = (const float*)d_in[18];
    const float* ln1_b = (const float*)d_in[19];
    const float* W_ff1 = (const float*)d_in[20];
    const float* b_ff1 = (const float*)d_in[21];
    const float* W_ff2 = (const float*)d_in[22];
    const float* b_ff2 = (const float*)d_in[23];
    const float* ln2_g = (const float*)d_in[24];
    const float* ln2_b = (const float*)d_in[25];
    const float* deconv_w = (const float*)d_in[26];
    const float* gn1_g = (const float*)d_in[27];
    const float* gn1_b = (const float*)d_in[28];
    const float* conv3_w = (const float*)d_in[29];
    const float* gn2_g = (const float*)d_in[30];
    const float* gn2_b = (const float*)d_in[31];

    float* outx = (float*)d_out;
    float* camx_out = outx + XOUT_;

    size_t woff = 0;
    auto alloc = [&](size_t bytes) -> void* {
        void* p = (char*)d_ws + woff;
        woff += (bytes + 255) & ~(size_t)255;
        return p;
    };
    unsigned short* full_bf = (unsigned short*)alloc((size_t)CAMS_ * LPIX_ * INC_ * 2);
    unsigned short* wtc_bf  = (unsigned short*)alloc((size_t)D_ * INC_ * 2);
    float*          feat_f  = (float*)alloc((size_t)CAMS_ * LPIX_ * D_ * 4);
    unsigned short* feat_bf = (unsigned short*)alloc((size_t)CAMS_ * LPIX_ * D_ * 2);
    float*          q_f     = (float*)alloc((size_t)NQ_ * D_ * 4);
    unsigned short* q_bf    = (unsigned short*)alloc((size_t)NQ_ * D_ * 2);
    float*          tmp_f   = (float*)alloc((size_t)NQ_ * D_ * 4);
    float*          off_f   = (float*)alloc((size_t)NQ_ * 192 * 4);
    float*          att_f   = (float*)alloc((size_t)NQ_ * 96 * 4);
    float*          val_f   = (float*)alloc((size_t)CAMS_ * LPIX_ * D_ * 4);
    unsigned short* agg_bf  = (unsigned short*)alloc((size_t)NQ_ * D_ * 2);
    unsigned short* h1_bf   = (unsigned short*)alloc((size_t)NQ_ * FFN_ * 2);
    unsigned short* wpack   = (unsigned short*)alloc((size_t)3 * 466944 * 2);
    unsigned short* wdec    = (unsigned short*)alloc((size_t)8 * 256 * 256 * 2);
    unsigned short* wcv     = (unsigned short*)alloc((size_t)27 * 192 * 256 * 2);
    unsigned short* x1_bf   = (unsigned short*)alloc((size_t)D_ * NVOX_ * 2);
    unsigned short* x1n_bf  = (unsigned short*)alloc((size_t)NVOX_ * D_ * 2);
    float*          gst     = (float*)alloc(64 * 4);

    auto pk = [&](int l, size_t o) { return wpack + (size_t)l * 466944 + o; };
    // per-layer pack offsets: off 0, att 49152, val 73728, out 139264, ff1 204800, ff2 335872

    // ---- weight packs ----
    k_cast_bf16<<<768, 256, 0, stream>>>(W_tc, wtc_bf, D_ * INC_);
    for (int l = 0; l < 3; ++l) {
        k_pack_bt<<<192, 256, 0, stream>>>(W_off + (size_t)l * 49152,  pk(l, 0),      256, 192);
        k_pack_bt<<< 96, 256, 0, stream>>>(W_att + (size_t)l * 24576,  pk(l, 49152),  256, 96);
        k_pack_bt<<<256, 256, 0, stream>>>(W_val + (size_t)l * 65536,  pk(l, 73728),  256, 256);
        k_pack_bt<<<256, 256, 0, stream>>>(W_out + (size_t)l * 65536,  pk(l, 139264), 256, 256);
        k_pack_bt<<<512, 256, 0, stream>>>(W_ff1 + (size_t)l * 131072, pk(l, 204800), 256, 512);
        k_pack_bt<<<512, 256, 0, stream>>>(W_ff2 + (size_t)l * 131072, pk(l, 335872), 512, 256);
    }
    k_pack_deconv<<<2048, 256, 0, stream>>>(deconv_w, wdec);
    k_pack_conv3<<<5184, 256, 0, stream>>>(conv3_w, wcv);

    // ---- restore + cam_x output + projection GEMM ----
    k_restore<<<3168, 256, 0, stream>>>(camera_x, ids, mask_token, full_bf, camx_out);
    k_gemm<<<dim3(33, 8), 32, 0, stream>>>(full_bf, wtc_bf, b_tc, nullptr,
        feat_f, feat_bf, cams_embeds, level_embeds,
        CAMS_ * LPIX_, INC_, D_, F_RELU | F_EMB | F_WF32 | F_WBF16);
    k_castq<<<20000, 256, 0, stream>>>(vol_emb, q_f, q_bf, NQ_ * D_);

    // ---- decoder layers ----
    for (int l = 0; l < 3; ++l) {
        k_gemm<<<dim3(33, 8), 32, 0, stream>>>(feat_bf, pk(l, 73728), b_val + (size_t)l * 256,
            nullptr, val_f, nullptr, nullptr, nullptr, CAMS_ * LPIX_, 256, 256, F_WF32);
        k_gemm<<<dim3(625, 6), 32, 0, stream>>>(q_bf, pk(l, 0), b_off + (size_t)l * 192,
            nullptr, off_f, nullptr, nullptr, nullptr, NQ_, 256, 192, F_WF32);
        k_gemm<<<dim3(625, 3), 32, 0, stream>>>(q_bf, pk(l, 49152), b_att + (size_t)l * 96,
            nullptr, att_f, nullptr, nullptr, nullptr, NQ_, 256, 96, F_WF32);
        k_softmax12<<<625, 256, 0, stream>>>(att_f);
        k_sample<<<dim3(NQ_, 8), 32, 0, stream>>>(off_f, att_f, val_f, agg_bf);
        k_gemm<<<dim3(625, 8), 32, 0, stream>>>(agg_bf, pk(l, 139264), b_out + (size_t)l * 256,
            q_f, tmp_f, nullptr, nullptr, nullptr, NQ_, 256, 256, F_WF32);
        k_layernorm<<<NQ_, 256, 0, stream>>>(tmp_f, ln1_g + (size_t)l * 256,
            ln1_b + (size_t)l * 256, q_f, q_bf);
        k_gemm<<<dim3(625, 16), 32, 0, stream>>>(q_bf, pk(l, 204800), b_ff1 + (size_t)l * 512,
            nullptr, nullptr, h1_bf, nullptr, nullptr, NQ_, 256, 512, F_RELU | F_WBF16);
        k_gemm<<<dim3(625, 8), 32, 0, stream>>>(h1_bf, pk(l, 335872), b_ff2 + (size_t)l * 256,
            q_f, tmp_f, nullptr, nullptr, nullptr, NQ_, 512, 256, F_WF32);
        k_layernorm<<<NQ_, 256, 0, stream>>>(tmp_f, ln2_g + (size_t)l * 256,
            ln2_b + (size_t)l * 256, q_f, q_bf);
    }

    // ---- transposed conv (8 tap GEMMs) + GN1 + relu ----
    k_deconv<<<dim3(1250, 16, 8), 32, 0, stream>>>(q_bf, wdec, x1_bf);
    k_zero<<<1, 64, 0, stream>>>(gst, 64);
    k_gnstats_bf16<<<dim3(256, 16), 256, 0, stream>>>(x1_bf, gst, 16, (long)NVOX_);
    k_gn1norm<<<160000, 256, 0, stream>>>(x1_bf, gst, gn1_g, gn1_b, x1n_bf);

    // ---- 3x3x3 conv -> d_out, then GN2 + relu in place ----
    k_conv3<<<10000, 384, 0, stream>>>(x1n_bf, wcv, outx);
    k_gnstats_f32<<<dim3(256, 16), 256, 0, stream>>>(outx, gst + 32, 12, (long)NVOX_);
    k_gn2norm<<<120000, 256, 0, stream>>>(outx, gst + 32, gn2_g, gn2_b);
}
